// MinkowskiGraspNet_23450521436172
// MI455X (gfx1250) — matrix operand order repaired and checked by reading, not compiled, not run on hardware
//
#include <hip/hip_runtime.h>
#include <hip/hip_bf16.h>

// ---------------------------------------------------------------------------
// Problem constants (from reference setup_inputs)
// ---------------------------------------------------------------------------
#define B_    2
#define N_    8192
#define S_    1024
#define C_    512
#define NS_   16
#define K0_   515          // C + 3
#define K0P_  544          // padded to multiple of 32
#define BS_   (B_ * S_)    // 2048 seeds
#define NCOL_ (BS_ * NS_)  // 32768 MLP columns
#define RADIUS_F 0.05f
#define HMIN_F  (-0.02f)
#define HMAX_F  0.04f
#define EPS_F   1e-5f

typedef __attribute__((ext_vector_type(16))) __bf16 v16bf;
typedef __attribute__((ext_vector_type(8)))  float  v8f;

// ---------------------------------------------------------------------------
// WMMA helpers: fragment layouts, contiguous k = 16*half + s (wave32)
// ---------------------------------------------------------------------------
__device__ __forceinline__ v8f wmma_bf16(v16bf a, v16bf b, v8f c) {
  return __builtin_amdgcn_wmma_f32_16x16x32_bf16(
      /*neg_a=*/false, a, /*neg_b=*/false, b,
      /*c_mod=*/(short)0, c, /*reuse_a=*/false, /*reuse_b=*/false);
}

// A (16x32, 16-bit): lane<16 holds row M=lane, K = k0..k0+15;
// lane>=16 holds row M=lane-16, K = k0+16..k0+31.
__device__ __forceinline__ v16bf load_fragA(const __bf16* __restrict__ row,
                                            int k0, int half) {
  union { uint4 u[2]; v16bf v; } t;
  const uint4* p = reinterpret_cast<const uint4*>(row + k0 + half * 16);
  t.u[0] = p[0];      // 8 bf16 = 16B
  t.u[1] = p[1];      // +8 elements = +16B
  return t.v;
}

// B (32x16, 16-bit): lane<16 holds col N=lane, K = k0..k0+15;
// lane>=16 holds col N=lane-16, K = k0+16..k0+31.
__device__ __forceinline__ v16bf load_fragB(const __bf16* __restrict__ row,
                                            int k0, int half) {
  union { uint4 u[2]; v16bf v; } t;
  const uint4* p = reinterpret_cast<const uint4*>(row + k0 + half * 16);
  t.u[0] = p[0];
  t.u[1] = p[1];
  return t.v;
}

// ---------------------------------------------------------------------------
// Generic bf16 WMMA GEMM:  C[m,n] = sum_k A[m,k] * Bt[n,k]
//   A : (M, K) bf16 row-major (K contiguous)       -- weights
//   Bt: (N, K) bf16 row-major (K contiguous)       -- activations, transposed
//   C : (M, N) f32 row-major (N contiguous)
// One wave computes a 32x32 tile (2x2 WMMA register blocking).
// M, N multiples of 32; K multiple of 32.
// ---------------------------------------------------------------------------
__global__ __launch_bounds__(32)
void gemm_wmma_bf16(const __bf16* __restrict__ A, const __bf16* __restrict__ Bt,
                    float* __restrict__ C, int M, int N, int K) {
  const int n0   = blockIdx.x * 32;
  const int m0   = blockIdx.y * 32;
  const int lane = threadIdx.x & 31;
  const int half = lane >> 4;
  const int l16  = lane & 15;

  const __bf16* a0 = A  + (size_t)(m0 + l16) * K;
  const __bf16* a1 = a0 + (size_t)16 * K;
  const __bf16* b0 = Bt + (size_t)(n0 + l16) * K;
  const __bf16* b1 = b0 + (size_t)16 * K;

  v8f acc00 = {}; v8f acc01 = {}; v8f acc10 = {}; v8f acc11 = {};

  for (int k0 = 0; k0 < K; k0 += 32) {
    v16bf fa0 = load_fragA(a0, k0, half);
    v16bf fa1 = load_fragA(a1, k0, half);
    v16bf fb0 = load_fragB(b0, k0, half);
    v16bf fb1 = load_fragB(b1, k0, half);
    acc00 = wmma_bf16(fa0, fb0, acc00);
    acc01 = wmma_bf16(fa0, fb1, acc01);
    acc10 = wmma_bf16(fa1, fb0, acc10);
    acc11 = wmma_bf16(fa1, fb1, acc11);
  }

  // D layout: VGPR r, lane -> (M = r + 8*half, N = l16)
  const int mr0 = m0 + half * 8;
  const int mr1 = m0 + 16 + half * 8;
#pragma unroll
  for (int r = 0; r < 8; ++r) {
    C[(size_t)(mr0 + r) * N + n0 + l16]      = acc00[r];
    C[(size_t)(mr0 + r) * N + n0 + 16 + l16] = acc01[r];
    C[(size_t)(mr1 + r) * N + n0 + l16]      = acc10[r];
    C[(size_t)(mr1 + r) * N + n0 + 16 + l16] = acc11[r];
  }
}

// ---------------------------------------------------------------------------
// Weight f32 -> bf16 with K padding (zeros beyond Cin)
// ---------------------------------------------------------------------------
__global__ __launch_bounds__(256)
void cvt_weight(const float* __restrict__ W, __bf16* __restrict__ Wb,
                int Cin, int Kpad) {
  const int o = blockIdx.x;
  for (int k = threadIdx.x; k < Kpad; k += blockDim.x) {
    float v = (k < Cin) ? W[(size_t)o * Cin + k] : 0.0f;
    Wb[(size_t)o * Kpad + k] = (__bf16)v;
  }
}

// ---------------------------------------------------------------------------
// Cylinder query (first NS_ valid indices in order) + rotated/normalized xyz
// rows (0..2) of X0t, plus zero-fill of the K padding rows. One wave per seed.
// ---------------------------------------------------------------------------
__global__ __launch_bounds__(32)
void cylinder_query(const float* __restrict__ xyz,
                    const float* __restrict__ new_xyz,
                    const float* __restrict__ rot,
                    int* __restrict__ idx_out,
                    __bf16* __restrict__ X0t) {
  const int seed = blockIdx.x;          // b*S + s
  const int b    = seed / S_;
  const int lane = threadIdx.x;

  const float cx = new_xyz[(size_t)seed * 3 + 0];
  const float cy = new_xyz[(size_t)seed * 3 + 1];
  const float cz = new_xyz[(size_t)seed * 3 + 2];
  const float* R = rot + (size_t)seed * 9;   // R[c][d]
  const float R00 = R[0], R01 = R[1], R02 = R[2];
  const float R10 = R[3], R11 = R[4], R12 = R[5];
  const float R20 = R[6], R21 = R[7], R22 = R[8];
  const float* pb = xyz + (size_t)b * N_ * 3;

  __shared__ int ssel[NS_];
  int cnt = 0;

  for (int n0 = 0; n0 < N_ && cnt < NS_; n0 += 32) {
    const int n = n0 + lane;
    const float dx = pb[n * 3 + 0] - cx;
    const float dy = pb[n * 3 + 1] - cy;
    const float dz = pb[n * 3 + 2] - cz;
    const float rx = dx * R00 + dy * R10 + dz * R20;
    const float ry = dx * R01 + dy * R11 + dz * R21;
    const float rz = dx * R02 + dy * R12 + dz * R22;
    const bool ok = (ry * ry + rz * rz < RADIUS_F * RADIUS_F) &&
                    (rx >= HMIN_F) && (rx <= HMAX_F);
    unsigned mask = __builtin_amdgcn_ballot_w32(ok);  // uniform across wave
    while (mask && cnt < NS_) {
      const int t = __builtin_ffs((int)mask) - 1;
      ssel[cnt] = n0 + t;                 // all lanes write same value
      ++cnt;
      mask &= mask - 1;
    }
  }
  __syncthreads();

  const int first = (cnt > 0) ? ssel[0] : 0;
  if (lane < NS_) {
    const int v = (lane < cnt) ? ssel[lane] : first;
    idx_out[(size_t)seed * NS_ + lane] = v;
    // rotated, normalized relative coords -> rows 0..2 of column (seed*16+lane)
    const float dx = pb[v * 3 + 0] - cx;
    const float dy = pb[v * 3 + 1] - cy;
    const float dz = pb[v * 3 + 2] - cz;
    const float gx = (dx * R00 + dy * R10 + dz * R20) * (1.0f / RADIUS_F);
    const float gy = (dx * R01 + dy * R11 + dz * R21) * (1.0f / RADIUS_F);
    const float gz = (dx * R02 + dy * R12 + dz * R22) * (1.0f / RADIUS_F);
    __bf16* col = X0t + (size_t)(seed * NS_ + lane) * K0P_;
    col[0] = (__bf16)gx;
    col[1] = (__bf16)gy;
    col[2] = (__bf16)gz;
    // zero the K padding rows (515..543)
    for (int c = K0_; c < K0P_; ++c) col[c] = (__bf16)0.0f;
  }
}

// ---------------------------------------------------------------------------
// Gather features into rows 3..514 of X0t (one block per column)
// ---------------------------------------------------------------------------
__global__ __launch_bounds__(256)
void gather_features(const float* __restrict__ feat, const int* __restrict__ idx,
                     __bf16* __restrict__ X0t) {
  const int col  = blockIdx.x;       // (b*S + s)*16 + k
  const int seed = col >> 4;
  const int b    = seed / S_;
  const int iv   = idx[col];
  const float* f = feat + (size_t)b * C_ * N_ + iv;
  __bf16* dst = X0t + (size_t)col * K0P_ + 3;
  for (int c = threadIdx.x; c < C_; c += blockDim.x)
    dst[c] = (__bf16)f[(size_t)c * N_];
}

// ---------------------------------------------------------------------------
// Per-channel training-mode BN stats (biased var): mean[m], rstd[m]
// ---------------------------------------------------------------------------
__global__ __launch_bounds__(256)
void bn_stats(const float* __restrict__ Y, float* __restrict__ mean,
              float* __restrict__ rstd, int Ncols) {
  const int m = blockIdx.x;
  const float* row = Y + (size_t)m * Ncols;
  float s = 0.0f, s2 = 0.0f;
  for (int i = threadIdx.x; i < Ncols; i += blockDim.x) {
    const float v = row[i];
    s += v; s2 += v * v;
  }
  __shared__ float sh0[256];
  __shared__ float sh1[256];
  sh0[threadIdx.x] = s; sh1[threadIdx.x] = s2;
  __syncthreads();
  for (int o = blockDim.x >> 1; o > 0; o >>= 1) {
    if (threadIdx.x < (unsigned)o) {
      sh0[threadIdx.x] += sh0[threadIdx.x + o];
      sh1[threadIdx.x] += sh1[threadIdx.x + o];
    }
    __syncthreads();
  }
  if (threadIdx.x == 0) {
    const float inv = 1.0f / (float)Ncols;
    const float mu  = sh0[0] * inv;
    const float var = sh1[0] * inv - mu * mu;
    mean[m] = mu;
    rstd[m] = rsqrtf(var + EPS_F);
  }
}

// ---------------------------------------------------------------------------
// normalize + affine + ReLU, write next activation matrix transposed (bf16)
// Xt[col*M + m] = relu(g*(Y[m,col]-mu)*rstd + beta)
// ---------------------------------------------------------------------------
__global__ __launch_bounds__(256)
void bn_relu_transpose(const float* __restrict__ Y, const float* __restrict__ mean,
                       const float* __restrict__ rstd, const float* __restrict__ gamma,
                       const float* __restrict__ beta, __bf16* __restrict__ Xt,
                       int M, int Ncols) {
  const int col = blockIdx.x;
  for (int m = threadIdx.x; m < M; m += blockDim.x) {
    float v = (Y[(size_t)m * Ncols + col] - mean[m]) * rstd[m] * gamma[m] + beta[m];
    v = fmaxf(v, 0.0f);
    Xt[(size_t)col * M + m] = (__bf16)v;
  }
}

// ---------------------------------------------------------------------------
// Layer-2 epilogue: BN + ReLU per element, max over ns=16 group, transpose
// Pt[col2*M + m] = max_k relu(bn(Y[m, col2*16+k]))
// ---------------------------------------------------------------------------
__global__ __launch_bounds__(256)
void bn_relu_maxpool_transpose(const float* __restrict__ Y,
                               const float* __restrict__ mean,
                               const float* __restrict__ rstd,
                               const float* __restrict__ gamma,
                               const float* __restrict__ beta,
                               __bf16* __restrict__ Pt, int M, int Ncols) {
  const int col2 = blockIdx.x;            // b*S + s
  for (int m = threadIdx.x; m < M; m += blockDim.x) {
    const float mu = mean[m], rs = rstd[m], g = gamma[m], be = beta[m];
    const float* row = Y + (size_t)m * Ncols + (size_t)col2 * NS_;
    float mx = 0.0f;                       // relu output lower bound
#pragma unroll
    for (int k = 0; k < NS_; ++k) {
      const float v = fmaxf((row[k] - mu) * rs * g + be, 0.0f);
      mx = fmaxf(mx, v);
    }
    Pt[(size_t)col2 * M + m] = (__bf16)mx;
  }
}

// ---------------------------------------------------------------------------
// Final: out[(b*S+s)*96 + o] = Y[o, col] + bc3[o]
// ---------------------------------------------------------------------------
__global__ __launch_bounds__(96)
void add_bias_transpose_out(const float* __restrict__ Y, const float* __restrict__ bc3,
                            float* __restrict__ out, int Mout, int Ncols) {
  const int col = blockIdx.x;
  for (int o = threadIdx.x; o < Mout; o += blockDim.x)
    out[(size_t)col * Mout + o] = Y[(size_t)o * Ncols + col] + bc3[o];
}

// ---------------------------------------------------------------------------
// Host launcher
// ---------------------------------------------------------------------------
extern "C" void kernel_launch(void* const* d_in, const int* in_sizes, int n_in,
                              void* d_out, int out_size, void* d_ws, size_t ws_size,
                              hipStream_t stream) {
  (void)in_sizes; (void)n_in; (void)out_size; (void)ws_size;

  const float* xyz      = (const float*)d_in[0];
  const float* new_xyz  = (const float*)d_in[1];
  const float* view_rot = (const float*)d_in[2];
  const float* features = (const float*)d_in[3];
  const float* W0  = (const float*)d_in[4];
  const float* g0  = (const float*)d_in[5];
  const float* be0 = (const float*)d_in[6];
  const float* W1  = (const float*)d_in[7];
  const float* g1  = (const float*)d_in[8];
  const float* be1 = (const float*)d_in[9];
  const float* W2  = (const float*)d_in[10];
  const float* g2  = (const float*)d_in[11];
  const float* be2 = (const float*)d_in[12];
  const float* Wc1 = (const float*)d_in[13];
  // bc1 (d_in[14]) cancels under training-mode BN -> skipped
  const float* gc1  = (const float*)d_in[15];
  const float* bec1 = (const float*)d_in[16];
  const float* Wc2  = (const float*)d_in[17];
  // bc2 (d_in[18]) cancels under training-mode BN -> skipped
  const float* gc2  = (const float*)d_in[19];
  const float* bec2 = (const float*)d_in[20];
  const float* Wc3  = (const float*)d_in[21];
  const float* bc3  = (const float*)d_in[22];
  float* out = (float*)d_out;

  // ---- workspace layout (256B aligned) ----
  char* base = (char*)d_ws;
  size_t off = 0;
  auto alloc = [&](size_t bytes) -> char* {
    char* p = base + off;
    off += (bytes + 255) & ~(size_t)255;
    return p;
  };
  int*     idxbuf = (int*)    alloc((size_t)NCOL_ * 4);
  __bf16*  Wb0    = (__bf16*) alloc((size_t)512 * K0P_ * 2);
  __bf16*  Wb1    = (__bf16*) alloc((size_t)256 * 512 * 2);
  __bf16*  Wb2    = (__bf16*) alloc((size_t)256 * 256 * 2);
  __bf16*  Wc1b   = (__bf16*) alloc((size_t)256 * 256 * 2);
  __bf16*  Wc2b   = (__bf16*) alloc((size_t)256 * 256 * 2);
  __bf16*  Wc3b   = (__bf16*) alloc((size_t)96  * 256 * 2);
  float*   meanb  = (float*)  alloc((size_t)512 * 4);
  float*   rstdb  = (float*)  alloc((size_t)512 * 4);
  __bf16*  Xt     = (__bf16*) alloc((size_t)NCOL_ * K0P_ * 2); // reused X0t/X1t/X2t
  float*   Ybuf   = (float*)  alloc((size_t)512 * NCOL_ * 4);  // reused Y0/Y1/Y2/head
  __bf16*  Pt     = (__bf16*) alloc((size_t)BS_ * 256 * 2);    // pooled / head acts

  // ---- 1) weights -> bf16 (padded) ----
  hipLaunchKernelGGL(cvt_weight, dim3(512), dim3(256), 0, stream, W0,  Wb0,  K0_, K0P_);
  hipLaunchKernelGGL(cvt_weight, dim3(256), dim3(256), 0, stream, W1,  Wb1,  512, 512);
  hipLaunchKernelGGL(cvt_weight, dim3(256), dim3(256), 0, stream, W2,  Wb2,  256, 256);
  hipLaunchKernelGGL(cvt_weight, dim3(256), dim3(256), 0, stream, Wc1, Wc1b, 256, 256);
  hipLaunchKernelGGL(cvt_weight, dim3(256), dim3(256), 0, stream, Wc2, Wc2b, 256, 256);
  hipLaunchKernelGGL(cvt_weight, dim3(96),  dim3(256), 0, stream, Wc3, Wc3b, 256, 256);

  // ---- 2) cylinder query + xyz rows + pad rows of X0t ----
  hipLaunchKernelGGL(cylinder_query, dim3(BS_), dim3(32), 0, stream,
                     xyz, new_xyz, view_rot, idxbuf, Xt);

  // ---- 3) gather features into X0t ----
  hipLaunchKernelGGL(gather_features, dim3(NCOL_), dim3(256), 0, stream,
                     features, idxbuf, Xt);

  // ---- 4) MLP layer 0: (512 x 544) x (544 x 32768) ----
  hipLaunchKernelGGL(gemm_wmma_bf16, dim3(NCOL_ / 32, 512 / 32), dim3(32), 0, stream,
                     Wb0, Xt, Ybuf, 512, NCOL_, K0P_);
  hipLaunchKernelGGL(bn_stats, dim3(512), dim3(256), 0, stream, Ybuf, meanb, rstdb, NCOL_);
  hipLaunchKernelGGL(bn_relu_transpose, dim3(NCOL_), dim3(256), 0, stream,
                     Ybuf, meanb, rstdb, g0, be0, Xt, 512, NCOL_);

  // ---- 5) MLP layer 1: (256 x 512) ----
  hipLaunchKernelGGL(gemm_wmma_bf16, dim3(NCOL_ / 32, 256 / 32), dim3(32), 0, stream,
                     Wb1, Xt, Ybuf, 256, NCOL_, 512);
  hipLaunchKernelGGL(bn_stats, dim3(256), dim3(256), 0, stream, Ybuf, meanb, rstdb, NCOL_);
  hipLaunchKernelGGL(bn_relu_transpose, dim3(NCOL_), dim3(256), 0, stream,
                     Ybuf, meanb, rstdb, g1, be1, Xt, 256, NCOL_);

  // ---- 6) MLP layer 2: (256 x 256), then BN+ReLU+maxpool -> Pt ----
  hipLaunchKernelGGL(gemm_wmma_bf16, dim3(NCOL_ / 32, 256 / 32), dim3(32), 0, stream,
                     Wb2, Xt, Ybuf, 256, NCOL_, 256);
  hipLaunchKernelGGL(bn_stats, dim3(256), dim3(256), 0, stream, Ybuf, meanb, rstdb, NCOL_);
  hipLaunchKernelGGL(bn_relu_maxpool_transpose, dim3(BS_), dim3(256), 0, stream,
                     Ybuf, meanb, rstdb, g2, be2, Pt, 256, NCOL_);

  // ---- 7) head conv1 (bias cancels under BN) ----
  hipLaunchKernelGGL(gemm_wmma_bf16, dim3(BS_ / 32, 256 / 32), dim3(32), 0, stream,
                     Wc1b, Pt, Ybuf, 256, BS_, 256);
  hipLaunchKernelGGL(bn_stats, dim3(256), dim3(256), 0, stream, Ybuf, meanb, rstdb, BS_);
  hipLaunchKernelGGL(bn_relu_transpose, dim3(BS_), dim3(256), 0, stream,
                     Ybuf, meanb, rstdb, gc1, bec1, Pt, 256, BS_);

  // ---- 8) head conv2 ----
  hipLaunchKernelGGL(gemm_wmma_bf16, dim3(BS_ / 32, 256 / 32), dim3(32), 0, stream,
                     Wc2b, Pt, Ybuf, 256, BS_, 256);
  hipLaunchKernelGGL(bn_stats, dim3(256), dim3(256), 0, stream, Ybuf, meanb, rstdb, BS_);
  hipLaunchKernelGGL(bn_relu_transpose, dim3(BS_), dim3(256), 0, stream,
                     Ybuf, meanb, rstdb, gc2, bec2, Pt, 256, BS_);

  // ---- 9) head conv3 (96 outputs) + bias + transpose to (B,S,96) ----
  hipLaunchKernelGGL(gemm_wmma_bf16, dim3(BS_ / 32, 96 / 32), dim3(32), 0, stream,
                     Wc3b, Pt, Ybuf, 96, BS_, 256);
  hipLaunchKernelGGL(add_bias_transpose_out, dim3(BS_), dim3(96), 0, stream,
                     Ybuf, bc3, out, 96, BS_);
}
